// RNN_59390807769111
// MI455X (gfx1250) — compile-verified, matrix operand order checked by reading, not run on hardware
//
#include <hip/hip_runtime.h>
#include <hip/hip_bf16.h>

typedef __attribute__((ext_vector_type(16))) __bf16 v16bf;
typedef __attribute__((ext_vector_type(8)))  float  v8f;

#define LL 512
#define BB 128
#define DD 1024
#define HH 1024
#define GROUPS 8        // BB / 16 batch-row tiles
#define WGPG   16       // workgroups per batch group (split over H)
#define NTW    4        // N-tiles per WG (16 cols each) -> 64 cols per WG
#define WPK_ELEMS (2*1024*1024)                 // (D+H) x H bf16, fragment-packed
#define XPK_ELEMS ((size_t)LL * GROUPS * 16384) // L x B x D bf16, fragment-packed
#define HPK_ELEMS ((size_t)GROUPS * 2 * 16384)  // double-buffered h, fragment-packed

__device__ __forceinline__ unsigned short f2bf(float f) {
  unsigned int u = __float_as_uint(f);
  u += 0x7fffu + ((u >> 16) & 1u);   // round-to-nearest-even
  return (unsigned short)(u >> 16);
}
__device__ __forceinline__ unsigned int pack2(float lo, float hi) {
  return (unsigned int)f2bf(lo) | ((unsigned int)f2bf(hi) << 16);
}

__global__ void zero_flags(int* flags, int n) {
  int i = blockIdx.x * blockDim.x + threadIdx.x;
  if (i < n) flags[i] = 0;
}

// Fragment addressing (mirrors 16-bit WMMA operand layout):
//   element (frag, lane, j): k_local = (j&7) + (j&8 ? 16:0) + (lane&16 ? 8:0)
//   A-matrix: m = lane&15 ; B-matrix: n = lane&15.

__global__ void pack_weights(const float* __restrict__ Wi,
                             const float* __restrict__ Wh,
                             unsigned short* __restrict__ wpk) {
  int idx = blockIdx.x * blockDim.x + threadIdx.x;
  if (idx >= WPK_ELEMS) return;
  int j    = idx & 15;
  int le   = (idx >> 4) & 31;
  int f    = idx >> 9;
  int s    = f & 63;                    // K-step 0..63 (K = 2048)
  int wgnt = f >> 6;                    // column tile 0..63
  int n = wgnt * 16 + (le & 15);
  int k = s * 32 + (j & 7) + ((j & 8) ? 16 : 0) + ((le & 16) ? 8 : 0);
  float v = (k < DD) ? Wi[(size_t)n * DD + k] : Wh[(size_t)n * HH + (k - DD)];
  wpk[idx] = f2bf(v);
}

// h0 -> bf16 A-fragments into hpk slot 1 (read slot for t=0)
__global__ void pack_h0(const float* __restrict__ h0,
                        unsigned short* __restrict__ hpk) {
  int idx = blockIdx.x * blockDim.x + threadIdx.x;
  if (idx >= (int)HPK_ELEMS / 2) return;
  int g = idx >> 14;
  int r = idx & 16383;
  int j  = r & 15;
  int le = (r >> 4) & 31;
  int s  = r >> 9;
  int m  = le & 15;
  int k = s * 32 + (j & 7) + ((j & 8) ? 16 : 0) + ((le & 16) ? 8 : 0);
  hpk[(((size_t)g * 2 + 1) << 14) + r] = f2bf(h0[(size_t)(g * 16 + m) * HH + k]);
}

// x -> bf16 A-fragments (parallel over all timesteps; off the serial path)
__global__ void pack_x(const float* __restrict__ x,
                       unsigned short* __restrict__ xpk) {
  size_t p = (size_t)blockIdx.x * blockDim.x + threadIdx.x;  // packet = 16 elems
  if (p >= XPK_ELEMS / 16) return;
  size_t tg = p >> 10;                 // t*GROUPS + g
  int r  = (int)(p & 1023);            // (s, lane)
  int s  = r >> 5;
  int le = r & 31;
  int m  = le & 15;
  int kb = s * 32 + ((le & 16) ? 8 : 0);
  size_t t = tg >> 3, g = tg & 7;
  const float* row = x + ((t * BB + g * 16 + m) * (size_t)DD + kb);
  float4 a0 = *(const float4*)(row);
  float4 a1 = *(const float4*)(row + 4);
  float4 c0 = *(const float4*)(row + 16);
  float4 c1 = *(const float4*)(row + 20);
  uint4 lo, hi;
  lo.x = pack2(a0.x, a0.y); lo.y = pack2(a0.z, a0.w);
  lo.z = pack2(a1.x, a1.y); lo.w = pack2(a1.z, a1.w);
  hi.x = pack2(c0.x, c0.y); hi.y = pack2(c0.z, c0.w);
  hi.z = pack2(c1.x, c1.y); hi.w = pack2(c1.z, c1.w);
  unsigned short* dst = xpk + (tg << 14) + ((size_t)r << 4);
  *(uint4*)(dst + 0) = lo;
  *(uint4*)(dst + 8) = hi;
}

template <bool XPK>
__global__ void __launch_bounds__(256)
rnn_scan(const float* __restrict__ x, const unsigned short* __restrict__ xpk,
         unsigned short* __restrict__ hpk,
         const float* __restrict__ Wi_b, const float* __restrict__ Wh_b,
         const unsigned short* __restrict__ wpk,
         int* __restrict__ flags, float* __restrict__ out) {
  __shared__ __align__(32) float red[NTW][32][8];            // K-slice reduction, 4 KB
  __shared__ __align__(32) unsigned short ldsA[16 * 1024];   // x staging (fallback only)

  const int tid  = threadIdx.x;
  const int lane = tid & 31;
  const int wave = tid >> 5;
  const int nt   = wave & (NTW - 1);
  const int ksl  = wave >> 2;                   // 0: x-half of K, 1: h-half
  const int g    = blockIdx.x / WGPG;
  const int wg   = blockIdx.x % WGPG;
  const int b0   = g * 16;
  const int n    = (wg * NTW + nt) * 16 + (lane & 15);
  const int mhi  = (lane & 16) ? 8 : 0;

  const float cval = (ksl == 0) ? (Wi_b[n] + Wh_b[n]) : 0.f;  // bias folded into C
  v8f cinit;
#pragma unroll
  for (int r = 0; r < 8; ++r) cinit[r] = cval;
  v8f zinit;
#pragma unroll
  for (int r = 0; r < 8; ++r) zinit[r] = 0.f;

  // producer scatter coords for bf16 h-fragments (column n, rows mhi..mhi+7)
  const int s_h = n >> 5, kl = n & 31;
  const int jh  = (kl & 7) + ((kl & 16) ? 8 : 0);
  const int hih = (kl & 8) ? 16 : 0;

  const size_t wbase = ((((size_t)(wg * NTW + nt) * 64) + (size_t)ksl * 32) << 9) +
                       ((size_t)lane << 4);

  for (int t = 0; t < LL; ++t) {
    if (t > 0) {
      if (tid == 0) {
        while (__hip_atomic_load(&flags[(t - 1) * GROUPS + g], __ATOMIC_ACQUIRE,
                                 __HIP_MEMORY_SCOPE_AGENT) < WGPG)
          __builtin_amdgcn_s_sleep(1);
      }
      __syncthreads();
    }

    if (t + 1 < LL) {  // warm next timestep's x fragments (global_prefetch_b8)
      if constexpr (XPK)
        __builtin_prefetch((const char*)xpk +
                           ((((size_t)(t + 1) * GROUPS + g) << 15)) + (tid << 7), 0, 1);
      else
        __builtin_prefetch(x + ((size_t)(t + 1) * BB + b0 + (tid & 15)) * DD +
                               (size_t)(tid >> 4) * 64, 0, 1);
    }

    if constexpr (!XPK) {  // stage + convert x-half into LDS fragments
      __syncthreads();
      for (int p = tid; p < 1024; p += 256) {
        int s  = p >> 5;
        int le = p & 31;
        int m  = le & 15;
        int kb = s * 32 + ((le & 16) ? 8 : 0);
        const float* row = x + ((size_t)t * BB + (b0 + m)) * DD + kb;
        float4 a0 = *(const float4*)(row);
        float4 a1 = *(const float4*)(row + 4);
        float4 c0 = *(const float4*)(row + 16);
        float4 c1 = *(const float4*)(row + 20);
        uint4 lo, hi;
        lo.x = pack2(a0.x, a0.y); lo.y = pack2(a0.z, a0.w);
        lo.z = pack2(a1.x, a1.y); lo.w = pack2(a1.z, a1.w);
        hi.x = pack2(c0.x, c0.y); hi.y = pack2(c0.z, c0.w);
        hi.z = pack2(c1.x, c1.y); hi.w = pack2(c1.z, c1.w);
        *(uint4*)(&ldsA[p * 16 + 0]) = lo;
        *(uint4*)(&ldsA[p * 16 + 8]) = hi;
      }
      __syncthreads();
    }

    // two independent accumulators break the WMMA->WMMA RAW chain (XDL pipelining)
    v8f acc0 = cinit, acc1 = zinit;

    if (ksl == 0) {            // x-half of K
      if constexpr (XPK) {
        const unsigned short* ab =
            xpk + (((size_t)t * GROUPS + g) << 14) + ((size_t)lane << 4);
#pragma unroll 4
        for (int s = 0; s < 32; s += 2) {
          v16bf a0v = *(const v16bf*)(ab + ((size_t)s << 9));
          v16bf b0v = *(const v16bf*)(wpk + wbase + ((size_t)s << 9));
          v16bf a1v = *(const v16bf*)(ab + ((size_t)(s + 1) << 9));
          v16bf b1v = *(const v16bf*)(wpk + wbase + ((size_t)(s + 1) << 9));
          acc0 = __builtin_amdgcn_wmma_f32_16x16x32_bf16(
              false, a0v, false, b0v, (short)0, acc0, false, false);
          acc1 = __builtin_amdgcn_wmma_f32_16x16x32_bf16(
              false, a1v, false, b1v, (short)0, acc1, false, false);
        }
      } else {
#pragma unroll 4
        for (int s = 0; s < 32; s += 2) {
          v16bf a0v = *(const v16bf*)(&ldsA[(s * 32 + lane) << 4]);
          v16bf b0v = *(const v16bf*)(wpk + wbase + ((size_t)s << 9));
          v16bf a1v = *(const v16bf*)(&ldsA[((s + 1) * 32 + lane) << 4]);
          v16bf b1v = *(const v16bf*)(wpk + wbase + ((size_t)(s + 1) << 9));
          acc0 = __builtin_amdgcn_wmma_f32_16x16x32_bf16(
              false, a0v, false, b0v, (short)0, acc0, false, false);
          acc1 = __builtin_amdgcn_wmma_f32_16x16x32_bf16(
              false, a1v, false, b1v, (short)0, acc1, false, false);
        }
      }
    } else {                   // h-half of K: ready-made bf16 fragments of h_{t-1}
      const unsigned short* ab =
          hpk + (((size_t)g * 2 + ((t + 1) & 1)) << 14) + ((size_t)lane << 4);
#pragma unroll 4
      for (int s = 0; s < 32; s += 2) {
        v16bf a0v = *(const v16bf*)(ab + ((size_t)s << 9));
        v16bf b0v = *(const v16bf*)(wpk + wbase + ((size_t)s << 9));
        v16bf a1v = *(const v16bf*)(ab + ((size_t)(s + 1) << 9));
        v16bf b1v = *(const v16bf*)(wpk + wbase + ((size_t)(s + 1) << 9));
        acc0 = __builtin_amdgcn_wmma_f32_16x16x32_bf16(
            false, a0v, false, b0v, (short)0, acc0, false, false);
        acc1 = __builtin_amdgcn_wmma_f32_16x16x32_bf16(
            false, a1v, false, b1v, (short)0, acc1, false, false);
      }
    }

    v8f acc = acc0 + acc1;

    // combine the two K-slices through LDS
    if (ksl == 1) *(v8f*)(&red[nt][lane][0]) = acc;   // ds_store_b128 x2
    __syncthreads();
    if (ksl == 0) {
      v8f other = *(const v8f*)(&red[nt][lane][0]);   // ds_load_b128 x2
      acc = acc + other;
      unsigned short* hb = hpk + (((size_t)g * 2 + (t & 1)) << 14);
#pragma unroll
      for (int r = 0; r < 8; ++r) {
        float v = tanhf(acc[r]);
        out[((size_t)t * BB + (b0 + mhi + r)) * HH + n] = v;            // fp32 output
        hb[((s_h * 32 + mhi + r + hih) << 4) + jh] = f2bf(v);           // bf16 frags
      }
    }
    __threadfence();
    __syncthreads();
    if (tid == 0)
      __hip_atomic_fetch_add(&flags[t * GROUPS + g], 1, __ATOMIC_RELEASE,
                             __HIP_MEMORY_SCOPE_AGENT);
  }
}

extern "C" void kernel_launch(void* const* d_in, const int* in_sizes, int n_in,
                              void* d_out, int out_size, void* d_ws, size_t ws_size,
                              hipStream_t stream) {
  (void)in_sizes; (void)n_in; (void)out_size;
  const float* x    = (const float*)d_in[0];
  const float* h0   = (const float*)d_in[1];
  const float* Wi_w = (const float*)d_in[2];
  const float* Wi_b = (const float*)d_in[3];
  const float* Wh_w = (const float*)d_in[4];
  const float* Wh_b = (const float*)d_in[5];
  float* out = (float*)d_out;

  const size_t xpkB = XPK_ELEMS * 2, hpkB = HPK_ELEMS * 2, wpkB = (size_t)WPK_ELEMS * 2;
  const size_t flagB = (size_t)LL * GROUPS * sizeof(int);
  const bool use_xpk = ws_size >= xpkB + hpkB + wpkB + flagB;

  char* wsp = (char*)d_ws;
  unsigned short* xpk = (unsigned short*)wsp;
  if (use_xpk) wsp += xpkB;
  unsigned short* hpk = (unsigned short*)wsp;  wsp += hpkB;
  unsigned short* wpk = (unsigned short*)wsp;  wsp += wpkB;
  int* flags = (int*)wsp;

  zero_flags<<<(LL * GROUPS + 255) / 256, 256, 0, stream>>>(flags, LL * GROUPS);
  pack_weights<<<WPK_ELEMS / 256, 256, 0, stream>>>(Wi_w, Wh_w, wpk);
  pack_h0<<<((int)HPK_ELEMS / 2 + 255) / 256, 256, 0, stream>>>(h0, hpk);
  if (use_xpk) {
    pack_x<<<(unsigned)((XPK_ELEMS / 16 + 255) / 256), 256, 0, stream>>>(x, xpk);
    rnn_scan<true><<<GROUPS * WGPG, 256, 0, stream>>>(x, xpk, hpk, Wi_b, Wh_b,
                                                      wpk, flags, out);
  } else {
    rnn_scan<false><<<GROUPS * WGPG, 256, 0, stream>>>(x, xpk, hpk, Wi_b, Wh_b,
                                                       wpk, flags, out);
  }
}
